// positionKPN_54829552500770
// MI455X (gfx1250) — compile-verified, matrix operand order checked by reading, not run on hardware
//
#include <hip/hip_runtime.h>
#include <hip/hip_bf16.h>

typedef __attribute__((ext_vector_type(16))) _Float16 v16h;
typedef __attribute__((ext_vector_type(8)))  _Float16 v8h;
typedef __attribute__((ext_vector_type(8)))  float    v8f;

#define KWMMA(a, b, acc) \
  __builtin_amdgcn_wmma_f32_16x16x32_f16(false, (a), false, (b), (short)0, (acc), false, false)

constexpr int C_IN  = 64;
constexpr int HIN   = 128;
constexpr int WIN   = 128;
constexpr int HO    = 256;
constexpr int WOUT  = 256;
constexpr int H2PAD = 264;   // 256 K-halves + 8 pad (bank stagger)
constexpr int WGPAD = 148;   // 144 rows + 4 pad floats

constexpr int W2_HALVES = 256 * 128;   // 32768  (16 Mtiles x 4 Ksteps x 512)
constexpr int W3_HALVES = 576 * 256;   // 147456 (36 Mtiles x 8 Ksteps x 512)

// ---------------------------------------------------------------------------
// Pre-pack: convert a row-major f32 weight matrix into f16 WMMA A-fragments,
// stored fragment-tile-major:  P[((mt*Ksteps + s)*32 + lane)*16 + j]
// gfx1250 16-bit A layout: lane m = lane&15, g = lane>>4,
//   half j -> K = 32*s + (j>>3)*16 + g*8 + (j&7).
// Each lane's 32 contiguous bytes are then one aligned v16h load in the GEMM.
// ---------------------------------------------------------------------------
__global__ __launch_bounds__(256)
void pack_wmma_frags(const float* __restrict__ W, _Float16* __restrict__ P,
                     int Ksteps, int Kdim, int total) {
  const int idx = blockIdx.x * 256 + threadIdx.x;
  if (idx >= total) return;
  const int j    = idx & 15;
  const int lane = (idx >> 4) & 31;
  const int fi   = idx >> 9;          // fragment-tile index = mt*Ksteps + s
  const int s    = fi % Ksteps;
  const int mt   = fi / Ksteps;
  const int row  = 16 * mt + (lane & 15);
  const int gg   = lane >> 4;
  const int K    = 32 * s + ((j >> 3) << 4) + (gg << 3) + (j & 7);
  P[idx] = (_Float16)W[row * Kdim + K];
}

__global__ __launch_bounds__(256)
void positionKPN_fused(const float*    __restrict__ xg,
                       const float*    __restrict__ pose,
                       const int*      __restrict__ interY,
                       const int*      __restrict__ interX,
                       const float*    __restrict__ W1,
                       const float*    __restrict__ b1,
                       const _Float16* __restrict__ W2p,
                       const float*    __restrict__ b2,
                       const _Float16* __restrict__ W3p,
                       const float*    __restrict__ b3,
                       float*          __restrict__ outg)
{
  __shared__ _Float16 sH2[128 * H2PAD];   // h2 activations, f16, [pixel][K]
  __shared__ float    sWgt[128 * WGPAD];  // predicted weights chunk, [pixel][16ch*9]
  __shared__ float    sW1[128 * 3];       // interleaved W1[k][0], W1[k][1], b1[k]

  const int tid  = threadIdx.x;
  const int nl   = tid & 15;          // lane % 16
  const int g    = (tid >> 4) & 1;    // lane group
  const int lane = tid & 31;
  const int wave = tid >> 5;          // 8 waves; wave == tile row (ly)
  const int pix  = wave * 16 + nl;    // this lane's pixel column in the tile

  if (tid < 128) {
    sW1[tid * 3 + 0] = W1[tid * 2 + 0];
    sW1[tid * 3 + 1] = W1[tid * 2 + 1];
    sW1[tid * 3 + 2] = b1[tid];
  }
  __syncthreads();

  const _Float16* w2l = W2p + lane * 16;   // this lane's slice of each fragment tile
  const _Float16* w3l = W3p + lane * 16;

  // ---------- layer 1: build h1 directly as layer-2 WMMA B fragments ----------
  {
    const int oy = blockIdx.y * 8 + wave;
    const int ox = blockIdx.x * 16 + nl;
    const float fx = pose[oy * WOUT + ox];
    const float fy = pose[HO * WOUT + oy * WOUT + ox];

    v16h Bp[4];
#pragma unroll
    for (int s = 0; s < 4; ++s) {
      v16h f;
#pragma unroll
      for (int j = 0; j < 16; ++j) {
        const int K = 32 * s + ((j >> 3) << 4) + (g << 3) + (j & 7);
        const float h = fmaf(fx, sW1[3 * K + 0], fmaf(fy, sW1[3 * K + 1], sW1[3 * K + 2]));
        f[j] = (_Float16)fmaxf(h, 0.0f);
      }
      Bp[s] = f;
    }

    // ---------- layer 2: h2 = relu(W2 @ h1 + b2) via WMMA, f16 -> LDS ----------
    for (int mt = 0; mt < 16; ++mt) {
      v8f acc = {0.f, 0.f, 0.f, 0.f, 0.f, 0.f, 0.f, 0.f};
#pragma unroll
      for (int s = 0; s < 4; ++s) {
        const v16h a = *(const v16h*)(w2l + (size_t)(mt * 4 + s) * 512);
        acc = KWMMA(a, Bp[s], acc);
      }
      const float4 bb0 = *(const float4*)(b2 + 16 * mt + 8 * g);
      const float4 bb1 = *(const float4*)(b2 + 16 * mt + 8 * g + 4);
      v8h hv;
      hv[0] = (_Float16)fmaxf(acc[0] + bb0.x, 0.f);
      hv[1] = (_Float16)fmaxf(acc[1] + bb0.y, 0.f);
      hv[2] = (_Float16)fmaxf(acc[2] + bb0.z, 0.f);
      hv[3] = (_Float16)fmaxf(acc[3] + bb0.w, 0.f);
      hv[4] = (_Float16)fmaxf(acc[4] + bb1.x, 0.f);
      hv[5] = (_Float16)fmaxf(acc[5] + bb1.y, 0.f);
      hv[6] = (_Float16)fmaxf(acc[6] + bb1.z, 0.f);
      hv[7] = (_Float16)fmaxf(acc[7] + bb1.w, 0.f);
      *(v8h*)(sH2 + pix * H2PAD + 16 * mt + 8 * g) = hv;   // ds_store_b128
    }
  }
  __syncthreads();

  // ---------- hold all 8 layer-3 B fragments (h2, K=256) in VGPRs ----------
  v16h B3[8];
#pragma unroll
  for (int s = 0; s < 8; ++s) {
    const v8h lo = *(const v8h*)(sH2 + pix * H2PAD + 32 * s + 8 * g);
    const v8h hi = *(const v8h*)(sH2 + pix * H2PAD + 32 * s + 16 + 8 * g);
    v16h f;
#pragma unroll
    for (int j = 0; j < 8; ++j) { f[j] = lo[j]; f[j + 8] = hi[j]; }
    B3[s] = f;
  }

  // ---------- per-pixel gather bookkeeping for the apply phase ----------
  const int ap  = tid & 127;
  const int sub = tid >> 7;
  const int aox = blockIdx.x * 16 + (ap & 15);
  const int aoy = blockIdx.y * 8  + (ap >> 4);
  const int iy = interY[aoy * WOUT + aox];
  const int ix = interX[aoy * WOUT + aox];
  int   offs[9];
  float okm[9];
#pragma unroll
  for (int k = 0; k < 9; ++k) {
    const int yy = iy + k / 3 - 1;          // pad = 1 (top/left), zero outside
    const int xx = ix + (k % 3) - 1;
    const bool ok = ((unsigned)yy < (unsigned)HIN) && ((unsigned)xx < (unsigned)WIN);
    offs[k] = ok ? (yy * WIN + xx) : 0;
    okm[k]  = ok ? 1.0f : 0.0f;
  }

  // ---------- layer 3 (16-channel chunks) + gathered 3x3 apply ----------
  for (int cc = 0; cc < 4; ++cc) {
    for (int mt = 0; mt < 9; ++mt) {
      const int fbase = (cc * 9 + mt) * 8;
      if (cc * 9 + mt < 35)   // prefetch next fragment block into WGP$/L2
        __builtin_prefetch((const void*)(w3l + (size_t)(fbase + 8) * 512), 0, 0);
      v8f acc = {0.f, 0.f, 0.f, 0.f, 0.f, 0.f, 0.f, 0.f};
#pragma unroll
      for (int s = 0; s < 8; ++s) {
        const v16h a = *(const v16h*)(w3l + (size_t)(fbase + s) * 512);
        acc = KWMMA(a, B3[s], acc);
      }
      const float4 bb0 = *(const float4*)(b3 + 144 * cc + 16 * mt + 8 * g);
      const float4 bb1 = *(const float4*)(b3 + 144 * cc + 16 * mt + 8 * g + 4);
      const float4 o0 = make_float4(acc[0] + bb0.x, acc[1] + bb0.y,
                                    acc[2] + bb0.z, acc[3] + bb0.w);
      const float4 o1 = make_float4(acc[4] + bb1.x, acc[5] + bb1.y,
                                    acc[6] + bb1.z, acc[7] + bb1.w);
      *(float4*)(sWgt + pix * WGPAD + 16 * mt + 8 * g)     = o0;
      *(float4*)(sWgt + pix * WGPAD + 16 * mt + 8 * g + 4) = o1;
    }
    __syncthreads();

    for (int c2 = 0; c2 < 8; ++c2) {
      const int cl = c2 * 2 + sub;
      const int c  = cc * 16 + cl;
      float wl[9];
#pragma unroll
      for (int k = 0; k < 9; ++k)
        wl[k] = sWgt[ap * WGPAD + 9 * cl + k] * okm[k];
#pragma unroll 2
      for (int b = 0; b < 8; ++b) {
        const float* xb = xg + (size_t)(b * C_IN + c) * (HIN * WIN);
        float sum = 0.f;
#pragma unroll
        for (int k = 0; k < 9; ++k)
          sum = fmaf(wl[k], xb[offs[k]], sum);
        outg[(size_t)(b * C_IN + c) * (HO * WOUT) + (size_t)aoy * WOUT + aox] = sum;
      }
    }
    __syncthreads();
  }
}

extern "C" void kernel_launch(void* const* d_in, const int* in_sizes, int n_in,
                              void* d_out, int out_size, void* d_ws, size_t ws_size,
                              hipStream_t stream) {
  const float* x    = (const float*)d_in[0];
  // d_in[1] = scale (compile-time constant 2, unused)
  const float* pose = (const float*)d_in[2];
  const int*   iy   = (const int*)d_in[3];
  const int*   ix   = (const int*)d_in[4];
  const float* W1   = (const float*)d_in[5];
  const float* b1   = (const float*)d_in[6];
  const float* W2   = (const float*)d_in[7];
  const float* b2   = (const float*)d_in[8];
  const float* W3   = (const float*)d_in[9];
  const float* b3   = (const float*)d_in[10];
  float* out = (float*)d_out;

  _Float16* W2p = (_Float16*)d_ws;            //  64 KB
  _Float16* W3p = W2p + W2_HALVES;            // 288 KB

  pack_wmma_frags<<<dim3((W2_HALVES + 255) / 256), dim3(256), 0, stream>>>(
      W2, W2p, /*Ksteps=*/4, /*Kdim=*/128, W2_HALVES);
  pack_wmma_frags<<<dim3((W3_HALVES + 255) / 256), dim3(256), 0, stream>>>(
      W3, W3p, /*Ksteps=*/8, /*Kdim=*/256, W3_HALVES);

  dim3 grid(WOUT / 16, HO / 8, 1);   // 16 x 32 = 512 blocks, 16x8 pixel tiles
  positionKPN_fused<<<grid, dim3(256, 1, 1), 0, stream>>>(
      x, pose, iy, ix, W1, b1, W2p, b2, W3p, b3, out);
}
